// tRNN_52063593563027
// MI455X (gfx1250) — matrix operand order repaired and checked by reading, not compiled, not run on hardware
//
#include <hip/hip_runtime.h>

typedef __attribute__((ext_vector_type(16))) _Float16 v16h;
typedef __attribute__((ext_vector_type(8)))  _Float16 v8h;
typedef __attribute__((ext_vector_type(4)))  _Float16 v4h;
typedef __attribute__((ext_vector_type(8)))  float    v8f;

union AFrag { v16h v; v8h h[2]; };

// Branchless tanh: native V_TANH_F32 on gfx1250 if the builtin exists,
// else (1-e)/(1+e) with e=exp(-2|x|) via fast exp + fast rcp (no EXEC branches).
__device__ __forceinline__ float fast_tanh(float x) {
#if defined(__has_builtin)
#if __has_builtin(__builtin_amdgcn_tanhf)
    return __builtin_amdgcn_tanhf(x);
#else
    float e = __expf(-2.0f * __builtin_fabsf(x));
    float t = (1.0f - e) * __builtin_amdgcn_rcpf(1.0f + e);
    return __builtin_copysignf(t, x);
#endif
#else
    float e = __expf(-2.0f * __builtin_fabsf(x));
    float t = (1.0f - e) * __builtin_amdgcn_rcpf(1.0f + e);
    return __builtin_copysignf(t, x);
#endif
}

// WMMA f16 16x16x32 layout (wave32):
//  A (16x32): lane L -> row M = L%16; two contiguous 8-half chunks per k-step:
//             K = kt*32 + 16*(L/16) (+0..7) and K = kt*32 + 16*(L/16) + 8 (+0..7)
//  B (32x16): lane L -> col N = L%16; half e -> K = kt*32 + 16*(L/16) + e
//  C/D f32  : vgpr r, lane L -> M = r + 8*(L/16), N = L%16

__global__ __launch_bounds__(256)
void trnn_kernel(const int* __restrict__ tokens,
                 const float* __restrict__ voc_w, const float* __restrict__ voc_b,
                 const float* __restrict__ cps_w, const float* __restrict__ cps_b,
                 const float* __restrict__ cpr_w, const float* __restrict__ cpr_b,
                 const float* __restrict__ sm_w,  const float* __restrict__ sm_b,
                 float* __restrict__ out)
{
    extern __shared__ char smem[];
    _Float16* bufA = (_Float16*)smem;                 // 256 rows x 128 halves (64 KB)
    _Float16* bufB = (_Float16*)(smem + 65536);       // 128 rows x 128 halves (32 KB)
    float*    zbuf = (float*)(smem + 65536);          // overlays bufB after tree pass

    const int tid  = threadIdx.x;
    const int lane = tid & 31;
    const int wave = tid >> 5;          // 8 waves, wave w owns output cols [16w,16w+16)
    const int hi   = lane >> 4;         // K-half selector
    const int nloc = lane & 15;
    const int ncol = (wave << 4) + nloc;     // 0..127
    const int b0   = blockIdx.x * 2;         // 2 batches (= 4 trees) per block

    // ---- resident B fragments for cps_w (reused by all 6 tree levels) ----
    v16h bw[8];
    {
        const float* wrow = cps_w + ncol * 256;   // cps_w[n][k], k contiguous
        #pragma unroll
        for (int kt = 0; kt < 8; ++kt) {
            const float4* w4 = (const float4*)(wrow + kt * 32 + hi * 16);
            #pragma unroll
            for (int q = 0; q < 4; ++q) {
                float4 f = w4[q];
                bw[kt][4*q+0] = (_Float16)f.x;
                bw[kt][4*q+1] = (_Float16)f.y;
                bw[kt][4*q+2] = (_Float16)f.z;
                bw[kt][4*q+3] = (_Float16)f.w;
            }
        }
    }
    const float bias_cps = cps_b[ncol];

    // ---- leaves: embedding gather voc_w[tok] + voc_b -> f16 rows in bufA ----
    // 4 trees x 64 leaves = 256 rows; tokens flat idx = b0*128 + row
    const int tokBase = b0 * 128;
    for (int c = tid; c < 256 * 32; c += 256) {   // 32 float4 chunks per row
        int row = c >> 5;
        int q   = c & 31;
        int tok = tokens[tokBase + row];
        float4 w = *(const float4*)(voc_w + tok * 128 + q * 4);
        float4 b = *(const float4*)(voc_b + q * 4);
        v4h h;
        h[0] = (_Float16)(w.x + b.x);
        h[1] = (_Float16)(w.y + b.y);
        h[2] = (_Float16)(w.z + b.z);
        h[3] = (_Float16)(w.w + b.w);
        *(v4h*)(bufA + row * 128 + q * 4) = h;
    }
    __syncthreads();

    // ---- 6 composition levels: parent row = contiguous 256-half child pair ----
    _Float16* cur = bufA;
    _Float16* nxt = bufB;
    int prows = 128;                       // parent rows at first level
    #pragma unroll 1
    for (int lvl = 0; lvl < 6; ++lvl) {
        const int mtiles = (prows + 15) >> 4;
        for (int mt = 0; mt < mtiles; ++mt) {
            const int pr = mt << 4;
            const _Float16* arow = cur + (pr + nloc) * 256;
            // Issue all 16 ds_load_b128 for this tile up front so the DS
            // pipeline stays full; in-order DS returns let the compiler use
            // staged s_wait_dscnt instead of waiting to zero per step.
            v8h a0[8], a1[8];
            #pragma unroll
            for (int kt = 0; kt < 8; ++kt) {
                a0[kt] = *(const v8h*)(arow + kt * 32 + hi * 16);
                a1[kt] = *(const v8h*)(arow + kt * 32 + hi * 16 + 8);
            }
            v8f acc;
            #pragma unroll
            for (int i = 0; i < 8; ++i) acc[i] = bias_cps;
            #pragma unroll
            for (int kt = 0; kt < 8; ++kt) {
                AFrag a; a.h[0] = a0[kt]; a.h[1] = a1[kt];
                acc = __builtin_amdgcn_wmma_f32_16x16x32_f16(
                          false, a.v, false, bw[kt], (short)0, acc, false, false);
            }
            #pragma unroll
            for (int r = 0; r < 8; ++r) {
                float t = fast_tanh(acc[r]);
                nxt[(pr + r + hi * 8) * 128 + ncol] = (_Float16)t;
            }
        }
        __syncthreads();
        _Float16* ts = cur; cur = nxt; nxt = ts;
        prows >>= 1;
    }
    // roots now in bufA: rows (b0:L, b0:R, b1:L, b1:R); pairs concat -> 2 rows x 256

    // ---- cpr layer: z = leaky_relu(root @ cpr_w^T + cpr_b), M=2 padded to 16 ----
    {
        v16h bwc[8];
        const float* wrow = cpr_w + ncol * 256;
        #pragma unroll
        for (int kt = 0; kt < 8; ++kt) {
            const float4* w4 = (const float4*)(wrow + kt * 32 + hi * 16);
            #pragma unroll
            for (int q = 0; q < 4; ++q) {
                float4 f = w4[q];
                bwc[kt][4*q+0] = (_Float16)f.x;
                bwc[kt][4*q+1] = (_Float16)f.y;
                bwc[kt][4*q+2] = (_Float16)f.z;
                bwc[kt][4*q+3] = (_Float16)f.w;
            }
        }
        const float bias = cpr_b[ncol];
        const _Float16* arow = cur + nloc * 256;
        v8h a0[8], a1[8];
        #pragma unroll
        for (int kt = 0; kt < 8; ++kt) {
            a0[kt] = *(const v8h*)(arow + kt * 32 + hi * 16);
            a1[kt] = *(const v8h*)(arow + kt * 32 + hi * 16 + 8);
        }
        v8f acc;
        #pragma unroll
        for (int i = 0; i < 8; ++i) acc[i] = bias;
        #pragma unroll
        for (int kt = 0; kt < 8; ++kt) {
            AFrag a; a.h[0] = a0[kt]; a.h[1] = a1[kt];
            acc = __builtin_amdgcn_wmma_f32_16x16x32_f16(
                      false, a.v, false, bwc[kt], (short)0, acc, false, false);
        }
        #pragma unroll
        for (int r = 0; r < 8; ++r) {
            float v = acc[r];
            v = v > 0.0f ? v : 0.01f * v;             // leaky_relu slope 0.01
            zbuf[(r + hi * 8) * 128 + ncol] = v;      // rows 0..1 valid
        }
    }
    __syncthreads();

    // ---- softmax head: 2 valid rows, 7 classes each (tiny; 2 threads) ----
    if (tid < 2) {
        const float* z = zbuf + tid * 128;
        float logit[7];
        float mx = -1e30f;
        for (int r = 0; r < 7; ++r) {
            float s = sm_b[r];
            const float* w = sm_w + r * 128;
            for (int c = 0; c < 128; ++c) s += z[c] * w[c];
            logit[r] = s;
            mx = fmaxf(mx, s);
        }
        float sum = 0.0f;
        for (int r = 0; r < 7; ++r) { float e = __expf(logit[r] - mx); logit[r] = e; sum += e; }
        float inv = 1.0f / sum;
        float* o = out + (b0 + tid) * 7;
        for (int r = 0; r < 7; ++r) o[r] = logit[r] * inv;
    }
}

extern "C" void kernel_launch(void* const* d_in, const int* in_sizes, int n_in,
                              void* d_out, int out_size, void* d_ws, size_t ws_size,
                              hipStream_t stream) {
    (void)n_in; (void)d_ws; (void)ws_size; (void)out_size;
    const int*   tokens = (const int*)  d_in[0];
    const float* voc_w  = (const float*)d_in[1];
    const float* voc_b  = (const float*)d_in[2];
    const float* cps_w  = (const float*)d_in[3];
    const float* cps_b  = (const float*)d_in[4];
    const float* cpr_w  = (const float*)d_in[5];
    const float* cpr_b  = (const float*)d_in[6];
    const float* sm_w   = (const float*)d_in[7];
    const float* sm_b   = (const float*)d_in[8];
    float* out = (float*)d_out;

    const int B = in_sizes[0] / (2 * 64);   // tokens is [B,2,64]
    const size_t smem = 65536 + 32768;      // bufA + bufB = 96 KB (<= 320 KB/WGP)
    static_cast<void>(hipFuncSetAttribute((const void*)trnn_kernel,
                      hipFuncAttributeMaxDynamicSharedMemorySize, (int)smem));
    trnn_kernel<<<dim3(B / 2), dim3(256), smem, stream>>>(
        tokens, voc_w, voc_b, cps_w, cps_b, cpr_w, cpr_b, sm_w, sm_b, out);
}
